// AttentionNet_26336739459245
// MI455X (gfx1250) — matrix operand order repaired and checked by reading, not compiled, not run on hardware
//
#include <hip/hip_runtime.h>

// ---------------------------------------------------------------------------
// Problem constants (from reference): B=8, T=512, J=64, H=200
// ---------------------------------------------------------------------------
constexpr int NB   = 8;            // batch
constexpr int NTT  = 512;          // time steps
constexpr int NJ   = 64;           // query length
constexpr int NH   = 200;          // hidden per direction
constexpr int ND2  = 2 * NH;       // 400
constexpr int NH4  = 4 * NH;       // 800 gate width
constexpr int NBT  = NB * NTT;     // 4096 rows
constexpr int KG   = 4 * ND2;      // 1600 (multiple of 64)
constexpr int KP1  = 448;          // 400 padded to multiple of 64
constexpr int KHH  = 224;          // 200 padded to multiple of 32 (LSTM kernel)

// GEMM block tiling: 20 wave32 waves, one 16x16 WMMA subtile each.
constexpr int BM  = 64;            // 4 M-subtiles
constexpr int BN  = 80;            // 5 N-subtiles (800 = 10*80, exact)
constexpr int BK2 = 64;            // staged K-slab = 2 WMMA k-steps per barrier pair

typedef __attribute__((ext_vector_type(16))) __bf16 v16bf;
typedef __attribute__((ext_vector_type(8)))  __bf16 v8bf;
typedef __attribute__((ext_vector_type(8)))  float  v8f;

__device__ __forceinline__ __bf16 f2bf(float f) {
  unsigned u = __float_as_uint(f);
  u += 0x7FFFu + ((u >> 16) & 1u);       // round-to-nearest-even
  unsigned short s = (unsigned short)(u >> 16);
  return __builtin_bit_cast(__bf16, s);
}

__device__ __forceinline__ float sigm(float x) { return 1.0f / (1.0f + expf(-x)); }

// ---------------------------------------------------------------------------
// fp32 -> bf16 with row padding (for weights): src rows x cols -> dst rows x dstride
// ---------------------------------------------------------------------------
__global__ void k_cvt_pad(const float* __restrict__ src, __bf16* __restrict__ dst,
                          int rows, int cols, int dstride) {
  int total = rows * dstride;
  for (int i = blockIdx.x * blockDim.x + threadIdx.x; i < total;
       i += gridDim.x * blockDim.x) {
    int r = i / dstride, c = i - r * dstride;
    dst[i] = (c < cols) ? f2bf(src[(size_t)r * cols + c]) : f2bf(0.0f);
  }
}

// ---------------------------------------------------------------------------
// Attention: per (b,t) compute S row over J, softmax, c2q row, and rowmax.
// Block = 64 threads (one per j).
// ---------------------------------------------------------------------------
__global__ void k_att_s(const float* __restrict__ ec, const float* __restrict__ eq,
                        const float* __restrict__ w_s,
                        float* __restrict__ c2q, float* __restrict__ rowmax) {
  int bt = blockIdx.x;
  int b = bt / NTT;
  int tid = threadIdx.x;
  __shared__ float ew3[ND2];
  __shared__ float red[64];
  __shared__ float sj[NJ];

  const float* er = ec + (size_t)bt * ND2;
  float part = 0.0f;
  for (int k = tid; k < ND2; k += 64) {
    float e = er[k];
    ew3[k] = e * w_s[2 * ND2 + k];
    part  += e * w_s[k];
  }
  red[tid] = part;
  __syncthreads();
  for (int s = 32; s > 0; s >>= 1) { if (tid < s) red[tid] += red[tid + s]; __syncthreads(); }
  float ecw1 = red[0];
  __syncthreads();

  const float* qr = eq + ((size_t)b * NJ + tid) * ND2;
  float sv = ecw1;
  for (int k = 0; k < ND2; ++k) {
    float q = qr[k];
    sv += q * w_s[ND2 + k] + q * ew3[k];
  }
  red[tid] = sv;
  __syncthreads();
  for (int s = 32; s > 0; s >>= 1) { if (tid < s) red[tid] = fmaxf(red[tid], red[tid + s]); __syncthreads(); }
  float m = red[0];
  if (tid == 0) rowmax[bt] = m;
  __syncthreads();
  float e = expf(sv - m);
  red[tid] = e;
  __syncthreads();
  for (int s = 32; s > 0; s >>= 1) { if (tid < s) red[tid] += red[tid + s]; __syncthreads(); }
  sj[tid] = e / red[0];
  __syncthreads();

  float* c2qr = c2q + (size_t)bt * ND2;
  for (int k = tid; k < ND2; k += 64) {
    float acc = 0.0f;
    const float* eqb = eq + (size_t)b * NJ * ND2 + k;
    for (int j = 0; j < NJ; ++j) acc += sj[j] * eqb[(size_t)j * ND2];
    c2qr[k] = acc;
  }
}

// ---------------------------------------------------------------------------
// Per batch: softmax_T(rowmax) -> b_att; q2c = b_att @ ec.  Block = 512 threads.
// ---------------------------------------------------------------------------
__global__ void k_q2c(const float* __restrict__ ec, const float* __restrict__ rowmax,
                      float* __restrict__ q2c) {
  int b = blockIdx.x;
  int t = threadIdx.x;
  __shared__ float att[NTT];
  __shared__ float red[NTT];
  float v = rowmax[b * NTT + t];
  red[t] = v;
  __syncthreads();
  for (int s = 256; s > 0; s >>= 1) { if (t < s) red[t] = fmaxf(red[t], red[t + s]); __syncthreads(); }
  float m = red[0];
  __syncthreads();
  float e = expf(v - m);
  red[t] = e;
  __syncthreads();
  for (int s = 256; s > 0; s >>= 1) { if (t < s) red[t] += red[t + s]; __syncthreads(); }
  att[t] = e / red[0];
  __syncthreads();
  for (int k = t; k < ND2; k += NTT) {
    float acc = 0.0f;
    for (int tt = 0; tt < NTT; ++tt) acc += att[tt] * ec[((size_t)b * NTT + tt) * ND2 + k];
    q2c[b * ND2 + k] = acc;
  }
}

// ---------------------------------------------------------------------------
// Assemble G = [ec, c2q, ec*c2q, ec*q2c] (fp32 + bf16 copy for the WMMA GEMM).
// ---------------------------------------------------------------------------
__global__ void k_G(const float* __restrict__ ec, const float* __restrict__ c2q,
                    const float* __restrict__ q2c,
                    float* __restrict__ G, __bf16* __restrict__ Gbf) {
  int bt = blockIdx.x;
  int b  = bt / NTT;
  const float* er = ec  + (size_t)bt * ND2;
  const float* cr = c2q + (size_t)bt * ND2;
  const float* qr = q2c + (size_t)b  * ND2;
  float*  g  = G   + (size_t)bt * KG;
  __bf16* gb = Gbf + (size_t)bt * KG;
  for (int k = threadIdx.x; k < ND2; k += blockDim.x) {
    float e = er[k], c = cr[k], q = qr[k];
    float v0 = e, v1 = c, v2 = e * c, v3 = e * q;
    g[k] = v0; g[ND2 + k] = v1; g[2 * ND2 + k] = v2; g[3 * ND2 + k] = v3;
    gb[k] = f2bf(v0); gb[ND2 + k] = f2bf(v1);
    gb[2 * ND2 + k] = f2bf(v2); gb[3 * ND2 + k] = f2bf(v3);
  }
}

// ---------------------------------------------------------------------------
// LDS-tiled bf16 WMMA GEMM: C[M,N] = A[M,K] * W[N,K]^T.
// A,W row-major, K contiguous, K % 64 == 0 (zero padded).
// Block = 640 threads (20 waves); block tile 64(M) x 80(N); per iteration the
// block stages a 64-wide K-slab (A 64x64 + B 80x64 = 18 KB LDS) and each wave
// runs TWO WMMAs per barrier pair.  ~0.45 KB global traffic per WMMA.
// ---------------------------------------------------------------------------
__global__ void k_gemm_bf16(const __bf16* __restrict__ A, const __bf16* __restrict__ W,
                            float* __restrict__ C, int M, int N, int K) {
  __shared__ __bf16 As[BM][BK2];   // 64x64 bf16 = 8 KB
  __shared__ __bf16 Bs[BN][BK2];   // 80x64 bf16 = 10 KB
  int tid  = threadIdx.x;
  int lane = tid & 31, wave = tid >> 5;       // waves 0..19
  int half = lane >> 4, r = lane & 15;
  int nblk = N / BN;
  int bm = blockIdx.x / nblk, bn = blockIdx.x - bm * nblk;
  if (bm * BM >= M) return;
  int ms = wave / 5, ns = wave - ms * 5;      // 4 x 5 wave grid

  const __bf16* Ab = A + ((size_t)bm * BM) * K;
  const __bf16* Wb = W + ((size_t)bn * BN) * K;

  constexpr int ACH = BM * BK2 / 8;           // 512 A chunks of 16 B
  constexpr int TCH = ACH + BN * BK2 / 8;     // + 640 B chunks = 1152

  v8f acc = {};
  for (int k0 = 0; k0 < K; k0 += BK2) {
    // prefetch next K-slab row heads (global_prefetch_b8)
    if (k0 + BK2 < K) {
      if (tid < BM)            __builtin_prefetch(Ab + (size_t)tid * K + k0 + BK2, 0, 1);
      else if (tid < BM + BN)  __builtin_prefetch(Wb + (size_t)(tid - BM) * K + k0 + BK2, 0, 1);
    }
    // stage A then B: 1152 x 16B chunks over 640 threads
    for (int c = tid; c < TCH; c += 640) {
      if (c < ACH) {
        int row = c >> 3, c8 = (c & 7) * 8;   // 8 chunks per 64-col row
        *(v8bf*)&As[row][c8] = *(const v8bf*)(Ab + (size_t)row * K + k0 + c8);
      } else {
        int j = c - ACH;
        int row = j >> 3, c8 = (j & 7) * 8;
        *(v8bf*)&Bs[row][c8] = *(const v8bf*)(Wb + (size_t)row * K + k0 + c8);
      }
    }
    __syncthreads();
#pragma unroll
    for (int kk = 0; kk < BK2; kk += 32) {
      v8bf a1 = *(const v8bf*)&As[ms * 16 + r][kk + half * 16];
      v8bf a2 = *(const v8bf*)&As[ms * 16 + r][kk + half * 16 + 8];
      v16bf av = __builtin_shufflevector(a1, a2, 0, 1, 2, 3, 4, 5, 6, 7,
                                                 8, 9, 10, 11, 12, 13, 14, 15);
      v16bf bv = *(const v16bf*)&Bs[ns * 16 + r][kk + half * 16];
      acc = __builtin_amdgcn_wmma_f32_16x16x32_bf16(false, av, false, bv,
                                                    (short)0, acc, false, false);
    }
    __syncthreads();
  }
  float* crow = C + ((size_t)(bm * BM + ms * 16 + half * 8)) * N + bn * BN + ns * 16 + r;
  for (int v = 0; v < 8; ++v) crow[(size_t)v * N] = acc[v];
}

// ---------------------------------------------------------------------------
// One BiLSTM layer, recurrent part.  grid = 2 (one block per direction),
// block = 512 (16 wave32 waves on one WGP).  Per step: gates(8x800) =
// hA(16x224,bf16 LDS) x Whh^T via 50 tiles x 7 WMMAs, + xproj + bias, then
// fused sigmoid/tanh state update; h kept in LDS as the next A operand.
// ---------------------------------------------------------------------------
__global__ void k_lstm(const float* __restrict__ xproj,   // (2, NBT, NH4)
                       const __bf16* __restrict__ Whb,    // (2, NH4, KHH) bf16 padded
                       const float* __restrict__ bias,    // (2, NH4)
                       float* __restrict__ hout,          // (B, T, 2H) fp32 concat
                       __bf16* hbf,                       // (B, T, KP1) bf16 or null
                       int hbfStride) {
  int dir = blockIdx.x;
  int tid = threadIdx.x;
  int lane = tid & 31, wave = tid >> 5;
  int half = lane >> 4, r = lane & 15;

  __shared__ __bf16 hA[16][KHH];   // A operand: rows 8..15, cols 200.. stay zero
  __shared__ float  cS[NB][NH];
  __shared__ float  gS[NB][NH4];

  for (int i = tid; i < 16 * KHH; i += 512) (&hA[0][0])[i] = f2bf(0.0f);
  for (int i = tid; i < NB * NH; i += 512)  (&cS[0][0])[i] = 0.0f;
  __syncthreads();

  const __bf16* Wd = Whb + (size_t)dir * NH4 * KHH;
  const float*  xd = xproj + (size_t)dir * NBT * NH4;
  const float*  bd = bias + dir * NH4;

  for (int step = 0; step < NTT; ++step) {
    int t = dir ? (NTT - 1 - step) : step;

    // ---- recurrent GEMM: gates = hA x Whh^T ----
    for (int tile = wave; tile < (NH4 >> 4); tile += 16) {
      const __bf16* wr = Wd + ((size_t)(tile * 16 + r)) * KHH + half * 16;
      const __bf16* ar = &hA[r][half * 16];
      v8f acc = {};
      for (int k0 = 0; k0 < KHH; k0 += 32) {
        v8bf a1 = *(const v8bf*)(ar + k0);
        v8bf a2 = *(const v8bf*)(ar + k0 + 8);
        v16bf av = __builtin_shufflevector(a1, a2, 0, 1, 2, 3, 4, 5, 6, 7,
                                                   8, 9, 10, 11, 12, 13, 14, 15);
        v16bf bv = *(const v16bf*)(wr + k0);
        acc = __builtin_amdgcn_wmma_f32_16x16x32_bf16(false, av, false, bv,
                                                      (short)0, acc, false, false);
      }
      if (half == 0) {                       // rows 0..7 = valid batches
        int n = tile * 16 + r;
        float bn = bd[n];
        for (int v = 0; v < NB; ++v)
          gS[v][n] = acc[v] + xd[((size_t)v * NTT + t) * NH4 + n] + bn;
      }
    }
    __syncthreads();

    // ---- fused gate nonlinearity + state update ----
    for (int idx = tid; idx < NB * NH; idx += 512) {
      int b = idx / NH, u = idx - b * NH;
      float gi = gS[b][u];
      float gf = gS[b][NH + u];
      float gg = gS[b][2 * NH + u];
      float go = gS[b][3 * NH + u];
      float cn = sigm(gf) * cS[b][u] + sigm(gi) * tanhf(gg);
      float hn = sigm(go) * tanhf(cn);
      cS[b][u] = cn;
      hA[b][u] = f2bf(hn);
      size_t o = (size_t)b * NTT + t;
      hout[o * ND2 + dir * NH + u] = hn;
      if (hbf) hbf[o * hbfStride + dir * NH + u] = f2bf(hn);
    }
    if (hbf && dir == 0) {                   // zero the K-padding columns
      for (int idx = tid; idx < NB * (KP1 - ND2); idx += 512) {
        int b = idx / (KP1 - ND2), u = ND2 + (idx - b * (KP1 - ND2));
        hbf[((size_t)b * NTT + t) * hbfStride + u] = f2bf(0.0f);
      }
    }
    __syncthreads();
  }
}

// ---------------------------------------------------------------------------
// logits[bt] = G[bt,:] . w[0:1600] + Mx[bt,:] . w[1600:2000] (+ bias scalar)
// ---------------------------------------------------------------------------
__global__ void k_dot(const float* __restrict__ Gm, const float* __restrict__ Mm,
                      const float* __restrict__ w, const float* __restrict__ bptr,
                      float* __restrict__ logits) {
  int bt = blockIdx.x;
  int tid = threadIdx.x;
  __shared__ float red[256];
  const float* gr = Gm + (size_t)bt * KG;
  const float* mr = Mm + (size_t)bt * ND2;
  float acc = 0.0f;
  for (int k = tid; k < KG;  k += 256) acc += gr[k] * w[k];
  for (int k = tid; k < ND2; k += 256) acc += mr[k] * w[KG + k];
  red[tid] = acc;
  __syncthreads();
  for (int s = 128; s > 0; s >>= 1) { if (tid < s) red[tid] += red[tid + s]; __syncthreads(); }
  if (tid == 0) logits[bt] = red[0] + (bptr ? bptr[0] : 0.0f);
}

// softmax over T per batch.  Block = 512.
__global__ void k_softmax_T(const float* __restrict__ logits, float* __restrict__ out) {
  int b = blockIdx.x, t = threadIdx.x;
  __shared__ float red[NTT];
  float v = logits[b * NTT + t];
  red[t] = v;
  __syncthreads();
  for (int s = 256; s > 0; s >>= 1) { if (t < s) red[t] = fmaxf(red[t], red[t + s]); __syncthreads(); }
  float m = red[0];
  __syncthreads();
  float e = expf(v - m);
  red[t] = e;
  __syncthreads();
  for (int s = 256; s > 0; s >>= 1) { if (t < s) red[t] += red[t + s]; __syncthreads(); }
  out[b * NTT + t] = e / red[0];
}

// ---------------------------------------------------------------------------
extern "C" void kernel_launch(void* const* d_in, const int* in_sizes, int n_in,
                              void* d_out, int out_size, void* d_ws, size_t ws_size,
                              hipStream_t stream) {
  (void)in_sizes; (void)n_in; (void)out_size; (void)ws_size;
  const float* ec   = (const float*)d_in[0];
  const float* eq   = (const float*)d_in[1];
  const float* w_s  = (const float*)d_in[2];
  const float* Wih0 = (const float*)d_in[3];
  const float* Whh0 = (const float*)d_in[4];
  const float* b0   = (const float*)d_in[5];
  const float* Wih1 = (const float*)d_in[6];
  const float* Whh1 = (const float*)d_in[7];
  const float* b1   = (const float*)d_in[8];
  const float* w_p1 = (const float*)d_in[9];
  const float* Wih2 = (const float*)d_in[10];
  const float* Whh2 = (const float*)d_in[11];
  const float* b2   = (const float*)d_in[12];
  const float* w_p2 = (const float*)d_in[13];
  const float* b_p2 = (const float*)d_in[14];
  float* out = (float*)d_out;

  char* ws = (char*)d_ws;
  size_t off = 0;
  auto alloc = [&](size_t bytes) -> void* {
    void* p = ws + off;
    off += (bytes + 255) & ~(size_t)255;
    return p;
  };
  float*  c2q     = (float*) alloc(sizeof(float)  * (size_t)NBT * ND2);
  float*  rowmax  = (float*) alloc(sizeof(float)  * NBT);
  float*  q2c     = (float*) alloc(sizeof(float)  * NB * ND2);
  float*  G       = (float*) alloc(sizeof(float)  * (size_t)NBT * KG);
  __bf16* Gbf     = (__bf16*)alloc(sizeof(__bf16) * (size_t)NBT * KG);
  float*  xproj   = (float*) alloc(sizeof(float)  * 2 * (size_t)NBT * NH4);
  float*  h0      = (float*) alloc(sizeof(float)  * (size_t)NBT * ND2);
  __bf16* h0bf    = (__bf16*)alloc(sizeof(__bf16) * (size_t)NBT * KP1);
  float*  M       = (float*) alloc(sizeof(float)  * (size_t)NBT * ND2);
  __bf16* Mbf     = (__bf16*)alloc(sizeof(__bf16) * (size_t)NBT * KP1);
  float*  M2      = (float*) alloc(sizeof(float)  * (size_t)NBT * ND2);
  float*  logits1 = (float*) alloc(sizeof(float)  * NBT);
  float*  logits2 = (float*) alloc(sizeof(float)  * NBT);
  __bf16* Wih0bf  = (__bf16*)alloc(sizeof(__bf16) * 2 * (size_t)NH4 * KG);
  __bf16* Wih1bf  = (__bf16*)alloc(sizeof(__bf16) * 2 * (size_t)NH4 * KP1);
  __bf16* Wih2bf  = (__bf16*)alloc(sizeof(__bf16) * 2 * (size_t)NH4 * KP1);
  __bf16* Whh0bf  = (__bf16*)alloc(sizeof(__bf16) * 2 * (size_t)NH4 * KHH);
  __bf16* Whh1bf  = (__bf16*)alloc(sizeof(__bf16) * 2 * (size_t)NH4 * KHH);
  __bf16* Whh2bf  = (__bf16*)alloc(sizeof(__bf16) * 2 * (size_t)NH4 * KHH);

  // --- weight precision conversion (bf16, zero-padded K) ---
  k_cvt_pad<<<512, 256, 0, stream>>>(Wih0, Wih0bf, 2 * NH4, KG,  KG);
  k_cvt_pad<<<256, 256, 0, stream>>>(Wih1, Wih1bf, 2 * NH4, ND2, KP1);
  k_cvt_pad<<<256, 256, 0, stream>>>(Wih2, Wih2bf, 2 * NH4, ND2, KP1);
  k_cvt_pad<<<128, 256, 0, stream>>>(Whh0, Whh0bf, 2 * NH4, NH,  KHH);
  k_cvt_pad<<<128, 256, 0, stream>>>(Whh1, Whh1bf, 2 * NH4, NH,  KHH);
  k_cvt_pad<<<128, 256, 0, stream>>>(Whh2, Whh2bf, 2 * NH4, NH,  KHH);

  // --- attention flow + G ---
  k_att_s<<<NBT, 64, 0, stream>>>(ec, eq, w_s, c2q, rowmax);
  k_q2c<<<NB, NTT, 0, stream>>>(ec, rowmax, q2c);
  k_G<<<NBT, 256, 0, stream>>>(ec, c2q, q2c, G, Gbf);

  const int gemmBlocks  = (NBT / BM) * (NH4 / BN);    // 64 * 10 = 640
  const int gemmThreads = (BM / 16) * (BN / 16) * 32; // 20 waves = 640

  // --- layer 0 ---
  for (int d = 0; d < 2; ++d)
    k_gemm_bf16<<<gemmBlocks, gemmThreads, 0, stream>>>(
        Gbf, Wih0bf + (size_t)d * NH4 * KG, xproj + (size_t)d * NBT * NH4,
        NBT, NH4, KG);
  k_lstm<<<2, 512, 0, stream>>>(xproj, Whh0bf, b0, h0, h0bf, KP1);

  // --- layer 1 ---
  for (int d = 0; d < 2; ++d)
    k_gemm_bf16<<<gemmBlocks, gemmThreads, 0, stream>>>(
        h0bf, Wih1bf + (size_t)d * NH4 * KP1, xproj + (size_t)d * NBT * NH4,
        NBT, NH4, KP1);
  k_lstm<<<2, 512, 0, stream>>>(xproj, Whh1bf, b1, M, Mbf, KP1);

  // --- p1 ---
  k_dot<<<NBT, 256, 0, stream>>>(G, M, w_p1, nullptr, logits1);
  k_softmax_T<<<NB, NTT, 0, stream>>>(logits1, out);

  // --- layer 2 ---
  for (int d = 0; d < 2; ++d)
    k_gemm_bf16<<<gemmBlocks, gemmThreads, 0, stream>>>(
        Mbf, Wih2bf + (size_t)d * NH4 * KP1, xproj + (size_t)d * NBT * NH4,
        NBT, NH4, KP1);
  k_lstm<<<2, 512, 0, stream>>>(xproj, Whh2bf, b2, M2, (__bf16*)nullptr, 0);

  // --- p2 ---
  k_dot<<<NBT, 256, 0, stream>>>(G, M2, w_p2, b_p2, logits2);
  k_softmax_T<<<NB, NTT, 0, stream>>>(logits2, out + NBT);
}
